// GenerativeODE_2199023256056
// MI455X (gfx1250) — matrix operand order repaired and checked by reading, not compiled, not run on hardware
//
#include <hip/hip_runtime.h>
#include <hip/hip_bf16.h>
#include <math.h>

#define DEVINL __device__ __forceinline__

typedef __bf16 bf16;
typedef __attribute__((ext_vector_type(16))) __bf16 v16bf;
typedef __attribute__((ext_vector_type(8)))  __bf16 v8bf;
typedef __attribute__((ext_vector_type(8)))  float  v8f;
typedef __attribute__((ext_vector_type(4)))  unsigned int v4u;
typedef __attribute__((ext_vector_type(8)))  unsigned int v8u;

// ---------------- problem constants ----------------
constexpr int B_    = 1024;
constexpr int T_    = 96;
constexpr int NZ_   = 1000;
constexpr int PF_   = 64, ZF_ = 32, Z_ = 64, P_ = 32;
constexpr int HH_   = 128, EH_ = 256, OH_ = 512, NB_ = 4;
constexpr int POS_  = Z_ + P_;           // 96
constexpr int SD_   = 2 * POS_;          // 192
constexpr int YD_   = SD_ + HH_;         // 320 (dW row stride)
constexpr int NET_IN_ = SD_ + HH_ + 2;   // 322
constexpr int KIN_  = 352;               // NET_IN padded to multiple of 32
constexpr float SIGMA_ = 0.1f;
constexpr int NPURP_ = 8;
constexpr float W24_ = 2.0f * 3.14159265358979323846f / 24.0f;
constexpr int NZP_ = 1008;               // zones padded to multiple of 16

constexpr size_t NROWS = (size_t)B_ * T_;

// ---------------- d_out layout (flat element offsets, reference return order) ----
constexpr size_t OFF_LOGITS = 0;
constexpr size_t OFF_LOC    = OFF_LOGITS + NROWS * NZ_;
constexpr size_t OFF_PURPL  = OFF_LOC    + NROWS * Z_;
constexpr size_t OFF_PURPF  = OFF_PURPL  + NROWS * NPURP_;
constexpr size_t OFF_P      = OFF_PURPF  + NROWS * P_;
constexpr size_t OFF_V      = OFF_P      + NROWS * POS_;
constexpr size_t OFF_MU     = OFF_V      + NROWS * POS_;
constexpr size_t OFF_LV     = OFF_MU     + (size_t)B_ * HH_;

// ---------------- workspace layout (bytes) ----------------
constexpr size_t WS_WNIN  = 0;                                            // bf16 [512][352]
constexpr size_t WS_WR1   = WS_WNIN  + sizeof(bf16) * 512 * KIN_;         // bf16 [4][512][512] (N-major)
constexpr size_t WS_WR2   = WS_WR1   + sizeof(bf16) * NB_ * OH_ * OH_;
constexpr size_t WS_WNOUT = WS_WR2   + sizeof(bf16) * NB_ * OH_ * OH_;    // bf16 [96][512]
constexpr size_t WS_CAND  = WS_WNOUT + sizeof(bf16) * POS_ * OH_;         // bf16 [1008][64]
constexpr size_t WS_STATE = WS_CAND  + sizeof(bf16) * (size_t)NZP_ * Z_;  // f32  [B][192]
constexpr size_t WS_HBF   = WS_STATE + sizeof(float) * (size_t)B_ * SD_;  // bf16 [B][128]

// ---------------- WMMA helpers ----------------
DEVINL v8f wmma_bf16(v16bf a, v16bf b, v8f c) {
  // D = A(16x32) * B(32x16) + C, fp32 accumulate, bf16 inputs
  return __builtin_amdgcn_wmma_f32_16x16x32_bf16(false, a, false, b, (short)0, c, false, false);
}

DEVINL v16bf join8(v8bf lo, v8bf hi) {
  v16bf r;
#pragma unroll
  for (int i = 0; i < 8; ++i) { r[i] = lo[i]; r[i + 8] = hi[i]; }
  return r;
}

// A fragment (16x32 bf16), rows m0..m0+15, K k0..k0+31 from row-major buffer.
// ISA layout: lanes 0-15 (m) hold K {k0..k0+7, k0+16..k0+23}; lanes 16-31 hold
// K {k0+8..k0+15, k0+24..k0+31}; two contiguous 16B chunks per lane.
DEVINL v16bf load_a_frag(const bf16* base, int ld, int m0, int k0, int lane) {
  int m  = m0 + (lane & 15);
  int hi = lane >> 4;
  const bf16* p = base + (size_t)m * ld + k0 + hi * 8;
  v8bf lo = *(const v8bf*)p;
  v8bf hh = *(const v8bf*)(p + 16);
  return join8(lo, hh);
}

// B fragment (32x16 bf16) from N-major weight WT[N][K]: lanes 0-15 (n) hold
// K {k0..k0+7, k0+16..k0+23}; lanes 16-31 hold K {k0+8..k0+15, k0+24..k0+31}.
DEVINL v16bf load_b_frag(const bf16* wt, int ld, int n0, int k0, int lane) {
  int n  = n0 + (lane & 15);
  int hi = lane >> 4;
  const bf16* p = wt + (size_t)n * ld + k0 + hi * 8;
  v8bf lo = *(const v8bf*)p;
  v8bf hh = *(const v8bf*)(p + 16);
  return join8(lo, hh);
}

// One 16-row tile strip: this wave owns 4 col tiles at n0 = wave*64 + cc*16.
template<int KSTEPS>
DEVINL void mm_tiles(const bf16* Abuf, int lda, const bf16* WT, int ldw,
                     int wave, int lane, v8f acc[4]) {
#pragma unroll 1
  for (int ks = 0; ks < KSTEPS; ++ks) {
    int k0 = ks * 32;
    v16bf a = load_a_frag(Abuf, lda, 0, k0, lane);
#pragma unroll
    for (int cc = 0; cc < 4; ++cc) {
      v16bf b = load_b_frag(WT, ldw, wave * 64 + cc * 16, k0, lane);
      acc[cc] = wmma_bf16(a, b, acc[cc]);
    }
  }
}

// D element (m,n): reg r, lane l -> m = ((l>>4)<<3)+r, n = (l&15)
DEVINL void store_tiles_act(bf16* Dbuf, const float* bias, int wave, int lane,
                            v8f acc[4], int act /*0=relu,1=tanh*/) {
  int nl = lane & 15;
  int mb = (lane >> 4) << 3;
#pragma unroll
  for (int cc = 0; cc < 4; ++cc) {
    int n = wave * 64 + cc * 16 + nl;
    float bv = bias[n];
#pragma unroll
    for (int r = 0; r < 8; ++r) {
      float v = acc[cc][r] + bv;
      v = (act == 0) ? fmaxf(v, 0.0f) : tanhf(v);
      Dbuf[(mb + r) * 512 + n] = (bf16)v;
    }
  }
}

DEVINL void store_tiles_residual(bf16* tbuf, const float* bias, int wave, int lane,
                                 v8f acc[4]) {
  int nl = lane & 15;
  int mb = (lane >> 4) << 3;
#pragma unroll
  for (int cc = 0; cc < 4; ++cc) {
    int n = wave * 64 + cc * 16 + nl;
    float bv = bias[n];
#pragma unroll
    for (int r = 0; r < 8; ++r) {
      int m = mb + r;
      float old = (float)tbuf[m * 512 + n];
      tbuf[m * 512 + n] = (bf16)tanhf(old + acc[cc][r] + bv);
    }
  }
}

// ---- Tensor Data Mover: DMA a 2D f32 tile (tile_dim0 x tile_dim1, row stride
// tensor_dim0_stride elements) from global memory into LDS.  D# built per
// CDNA5 ISA and issued once per workgroup (wave 0);
// completion tracked with TENSORcnt.
DEVINL void tdm_load_tile_f32(unsigned lds_addr, const float* gptr,
                              unsigned tile_d0, unsigned tile_d1,
                              unsigned tensor_d0, unsigned tensor_d1,
                              unsigned stride_d0) {
  unsigned long long ga = (unsigned long long)(uintptr_t)gptr;
  v4u g0;
  g0[0] = 1u;                                           // count=1, user D#
  g0[1] = lds_addr;                                     // lds_addr (bytes)
  g0[2] = (unsigned)(ga & 0xffffffffu);                 // global_addr[31:0]
  g0[3] = (unsigned)((ga >> 32) & 0x01ffffffu)          // global_addr[56:32]
        | 0x80000000u;                                  // type = 2 ("image")
  v8u g1;
  g1[0] = 0x00020000u;                                  // data_size = 2 (4B)
  g1[1] = (tensor_d0 & 0xffffu) << 16;                  // tensor_dim0[15:0]
  g1[2] = ((tensor_d0 >> 16) & 0xffffu)                 // tensor_dim0[31:16]
        | ((tensor_d1 & 0xffffu) << 16);                // tensor_dim1[15:0]
  g1[3] = ((tensor_d1 >> 16) & 0xffffu)                 // tensor_dim1[31:16]
        | ((tile_d0 & 0xffffu) << 16);                  // tile_dim0
  g1[4] = (tile_d1 & 0xffffu);                          // tile_dim1 (tile_dim2=0)
  g1[5] = stride_d0;                                    // tensor_dim0_stride[31:0]
  g1[6] = 0u;
  g1[7] = 0u;
  asm volatile("tensor_load_to_lds %0, %1" :: "s"(g0), "s"(g1) : "memory");
}

// ---------------- kernel 0: weight transpose + bf16 convert ----------------
__global__ __launch_bounds__(256) void k_prep_weights(
    const float* Wn_in, const float* Wr1, const float* Wr2, const float* Wn_out,
    bf16* WnInT, bf16* Wr1T, bf16* Wr2T, bf16* WnOutT) {
  const int n_in  = 512 * KIN_;
  const int n_r   = NB_ * OH_ * OH_;
  const int n_out = POS_ * OH_;
  const int total = n_in + 2 * n_r + n_out;
  for (int idx = blockIdx.x * blockDim.x + threadIdx.x; idx < total;
       idx += gridDim.x * blockDim.x) {
    if (idx < n_in) {
      int n = idx / KIN_, k = idx % KIN_;
      WnInT[idx] = (bf16)(k < NET_IN_ ? Wn_in[(size_t)k * OH_ + n] : 0.0f);
    } else if (idx < n_in + n_r) {
      int r = idx - n_in;
      int b = r / (OH_ * OH_), rr = r % (OH_ * OH_);
      int n = rr / OH_, k = rr % OH_;
      Wr1T[r] = (bf16)Wr1[((size_t)b * OH_ + k) * OH_ + n];
    } else if (idx < n_in + 2 * n_r) {
      int r = idx - n_in - n_r;
      int b = r / (OH_ * OH_), rr = r % (OH_ * OH_);
      int n = rr / OH_, k = rr % OH_;
      Wr2T[r] = (bf16)Wr2[((size_t)b * OH_ + k) * OH_ + n];
    } else {
      int r = idx - n_in - 2 * n_r;
      int n = r / OH_, k = r % OH_;
      WnOutT[r] = (bf16)Wn_out[(size_t)k * POS_ + n];
    }
  }
}

// ---------------- kernel 1: candidate zone embeddings (l2norm) -> bf16 -------
__global__ __launch_bounds__(64) void k_cand(const float* azf, const float* Wz,
                                             const float* bz, bf16* candT) {
  __shared__ float e[64];
  __shared__ float sscale;
  int n = blockIdx.x, tid = threadIdx.x;
  if (n >= NZ_) { candT[(size_t)n * Z_ + tid] = (bf16)0.0f; return; }
  float a = bz[tid];
  for (int k = 0; k < ZF_; ++k) a += azf[(size_t)n * ZF_ + k] * Wz[k * Z_ + tid];
  e[tid] = a;
  __syncthreads();
  if (tid == 0) {
    float s = 0.f;
    for (int k = 0; k < 64; ++k) s += e[k] * e[k];
    sscale = 1.0f / fmaxf(sqrtf(s), 1e-12f);
  }
  __syncthreads();
  candT[(size_t)n * Z_ + tid] = (bf16)(a * sscale);
}

// ---------------- kernel 2: encoder MLP + h0 + initial state -----------------
__global__ __launch_bounds__(256) void k_encoder(
    const float* pf, const float* hz, const float* wzf, const float* purp,
    const float* eps, const float* Wz, const float* bz,
    const float* We1, const float* be1, const float* We2, const float* be2,
    float* out_mu, float* out_lv, float* state, bf16* hbf) {
  __shared__ float s_in[224];
  __shared__ float s_hid[256];
  __shared__ float s_mu[128];
  __shared__ float s_lv[128];
  int b = blockIdx.x, tid = threadIdx.x;
  if (tid < 64) {
    s_in[tid] = pf[(size_t)b * PF_ + tid];
    float a = bz[tid];
    for (int k = 0; k < ZF_; ++k) a += hz[(size_t)b * ZF_ + k] * Wz[k * Z_ + tid];
    s_in[64 + tid] = a;                       // home embed
  } else if (tid < 128) {
    int n = tid - 64;
    float a = bz[n];
    for (int k = 0; k < ZF_; ++k) a += wzf[(size_t)b * ZF_ + k] * Wz[k * Z_ + n];
    s_in[128 + n] = a;                        // work embed
  } else if (tid < 160) {
    s_in[192 + (tid - 128)] = purp[(size_t)b * P_ + (tid - 128)];
  }
  __syncthreads();
  {
    float a = be1[tid];
    for (int k = 0; k < 224; ++k) a += s_in[k] * We1[(size_t)k * EH_ + tid];
    s_hid[tid] = fmaxf(a, 0.0f);
  }
  __syncthreads();
  {
    float a = be2[tid];
    for (int k = 0; k < 256; ++k) a += s_hid[k] * We2[(size_t)k * (2 * HH_) + tid];
    if (tid < 128) { s_mu[tid] = a; out_mu[(size_t)b * HH_ + tid] = a; }
    else           { s_lv[tid - 128] = a; out_lv[(size_t)b * HH_ + (tid - 128)] = a; }
  }
  __syncthreads();
  if (tid < 128) {
    float h0 = s_mu[tid] + expf(0.5f * s_lv[tid]) * eps[(size_t)b * HH_ + tid];
    hbf[(size_t)b * HH_ + tid] = (bf16)h0;
  }
  if (tid < SD_) {
    float v = (tid < 64) ? s_in[64 + tid]
            : (tid < 96) ? s_in[192 + (tid - 64)] : 0.0f;  // p0=[home,purp], v0=0
    state[(size_t)b * SD_ + tid] = v;
  }
}

// ---------------- kernel 3: persistent SDE scan (the hot loop) ---------------
DEVINL void write_outputs(const float* st, int rbase, int tcur, float* out_p,
                          float* out_v, float* out_loc, float* out_pf, int tid) {
  for (int idx = tid; idx < 16 * 192; idx += 256) {
    int i = idx / 192, j = idx % 192;
    float v = st[idx];
    size_t rt = (size_t)(rbase + i) * T_ + tcur;
    if (j < 96) {
      out_p[rt * 96 + j] = v;
      if (j < 64) out_loc[rt * 64 + j] = v;
      else        out_pf[rt * 32 + (j - 64)] = v;
    } else {
      out_v[rt * 96 + (j - 96)] = v;
    }
  }
}

__global__ __launch_bounds__(256) void k_scan(
    const float* times, const float* dW, const float* state_init, const bf16* hbf,
    const bf16* WnInT, const float* bn_in,
    const bf16* Wr1T, const float* br1,
    const bf16* Wr2T, const float* br2,
    const bf16* WnOutT, const float* bn_out,
    float* out_p, float* out_v, float* out_loc, float* out_pf) {
  __shared__ float st[16 * 192];    // [p,v] state, fp32
  __shared__ float nb[16 * 192];    // per-step Brownian noise tile (TDM target)
  __shared__ bf16  hb[16 * 128];    // constant latent h (zero drift & diffusion)
  __shared__ bf16  xb[16 * 512];    // activation ping
  __shared__ bf16  tb[16 * 512];    // activation pong (holds "x")

  const int tid = threadIdx.x, lane = tid & 31, wave = tid >> 5;
  const int rbase = blockIdx.x * 16;
  const unsigned nb_lds = (unsigned)(uintptr_t)&nb[0];

  for (int idx = tid; idx < 16 * 192; idx += 256)
    st[idx] = state_init[(size_t)(rbase + idx / 192) * SD_ + (idx % 192)];
  for (int idx = tid; idx < 16 * 128; idx += 256)
    hb[idx] = hbf[(size_t)(rbase + (idx >> 7)) * HH_ + (idx & 127)];
  __syncthreads();
  write_outputs(st, rbase, 0, out_p, out_v, out_loc, out_pf, tid);

  for (int t = 0; t < T_ - 1; ++t) {
    float t0 = times[t];
    float dt = times[t + 1] - t0;
    float sq = sqrtf(dt) * SIGMA_;
    float sv = sinf(t0 * W24_), cv = cosf(t0 * W24_);
    __syncthreads();   // nb free (last step's reads done), xb free
    // Kick off the async TDM fetch of this step's noise tile; it overlaps
    // with the ten GEMMs below and is waited on just before the Euler update.
    if (wave == 0) {
      tdm_load_tile_f32(nb_lds, dW + ((size_t)t * B_ + rbase) * YD_,
                        /*tile*/ SD_, 16, /*tensor*/ YD_, 16, /*stride*/ YD_);
    }
    // stage x_in = [p, v, h, sin, cos, pad] as bf16
    for (int idx = tid; idx < 16 * KIN_; idx += 256) {
      int i = idx / KIN_, j = idx % KIN_;
      if (j < SD_)            xb[i * 512 + j] = (bf16)st[i * 192 + j];
      else if (j < SD_ + HH_) xb[i * 512 + j] = hb[i * 128 + (j - SD_)];
      else if (j == SD_ + HH_)     xb[i * 512 + j] = (bf16)sv;
      else if (j == SD_ + HH_ + 1) xb[i * 512 + j] = (bf16)cv;
      else                    xb[i * 512 + j] = (bf16)0.0f;
    }
    __syncthreads();
    { // x = relu(x_in @ Wn_in + bn_in) -> tb
      v8f acc[4] = {};
      mm_tiles<11>(xb, 512, WnInT, KIN_, wave, lane, acc);
      store_tiles_act(tb, bn_in, wave, lane, acc, 0);
    }
    __syncthreads();
    for (int ib = 0; ib < NB_; ++ib) {   // x = tanh(x + tanh(x@Wr1+br1)@Wr2 + br2)
      {
        v8f acc[4] = {};
        mm_tiles<16>(tb, 512, Wr1T + (size_t)ib * OH_ * OH_, OH_, wave, lane, acc);
        store_tiles_act(xb, br1 + ib * OH_, wave, lane, acc, 1);
      }
      __syncthreads();
      {
        v8f acc[4] = {};
        mm_tiles<16>(xb, 512, Wr2T + (size_t)ib * OH_ * OH_, OH_, wave, lane, acc);
        store_tiles_residual(tb, br2 + ib * OH_, wave, lane, acc);
      }
      __syncthreads();
    }
    // dv = tanh(x @ Wn_out + bn_out): 6 col tiles -> waves 0..5
    v8f oacc = {};
    if (wave < 6) {
#pragma unroll 1
      for (int ks = 0; ks < 16; ++ks) {
        int k0 = ks * 32;
        v16bf a = load_a_frag(tb, 512, 0, k0, lane);
        v16bf b = load_b_frag(WnOutT, 512, wave * 16, k0, lane);
        oacc = wmma_bf16(a, b, oacc);
      }
    }
    if (wave == 0) __builtin_amdgcn_s_wait_tensorcnt(0);  // noise tile landed
    __syncthreads();
    // p += v*dt + sigma*sqrt(dt)*dw_p   (reads old v)
    for (int idx = tid; idx < 16 * 96; idx += 256) {
      int i = idx / 96, j = idx % 96;
      st[i * 192 + j] += st[i * 192 + 96 + j] * dt + sq * nb[i * 192 + j];
    }
    __syncthreads();
    // v += dv*dt + sigma*sqrt(dt)*dw_v
    if (wave < 6) {
      int n  = wave * 16 + (lane & 15);
      int mb = (lane >> 4) << 3;
      float bv = bn_out[n];
#pragma unroll
      for (int r = 0; r < 8; ++r) {
        int m = mb + r;
        float dv = tanhf(oacc[r] + bv);
        st[m * 192 + 96 + n] += dv * dt + sq * nb[m * 192 + 96 + n];
      }
    }
    __syncthreads();
    write_outputs(st, rbase, t + 1, out_p, out_v, out_loc, out_pf, tid);
  }
}

// ---------------- kernel 4: decoder (l2norm + WMMA logits GEMM) --------------
__global__ __launch_bounds__(256) void k_decoder(
    const float* predp, const float* Wdl, const float* bdl, const bf16* candT,
    const float* Wdp, const float* bdp, float* out_logits, float* out_purp) {
  __shared__ float s_pls[64 * 64];
  __shared__ float s_w[64 * 64];
  __shared__ float s_tg[64 * 64];
  __shared__ bf16  s_tgb[64 * 64];
  __shared__ float s_rs[64];
  const int tid = threadIdx.x, lane = tid & 31, wave = tid >> 5;
  const size_t row0 = (size_t)blockIdx.x * 64;

  for (int idx = tid; idx < 64 * 64; idx += 256) {
    int i = idx >> 6, k = idx & 63;
    s_pls[idx] = predp[(row0 + i) * 96 + k];   // loc embed = pred_p[..,:64]
    s_w[idx]   = Wdl[idx];
  }
  __syncthreads();
  for (int idx = tid; idx < 64 * 64; idx += 256) {
    int i = idx >> 6, n = idx & 63;
    float a = bdl[n];
#pragma unroll 4
    for (int k = 0; k < 64; ++k) a += s_pls[i * 64 + k] * s_w[k * 64 + n];
    s_tg[idx] = a;
  }
  __syncthreads();
  if (tid < 64) {
    float ss = 0.f;
    for (int k = 0; k < 64; ++k) { float v = s_tg[tid * 64 + k]; ss += v * v; }
    s_rs[tid] = 1.0f / fmaxf(sqrtf(ss), 1e-12f);
  }
  __syncthreads();
  for (int idx = tid; idx < 64 * 64; idx += 256)
    s_tgb[idx] = (bf16)(s_tg[idx] * s_rs[idx >> 6]);
  __syncthreads();
  // logits(64 x 1008) = tgt(64x64) @ cand^T, cand stored [n][64] bf16
  for (int ct = wave; ct < NZP_ / 16; ct += 8) {
#pragma unroll
    for (int rt = 0; rt < 4; ++rt) {
      v8f acc = {};
#pragma unroll
      for (int ks = 0; ks < 2; ++ks) {
        int k0 = ks * 32;
        v16bf a = load_a_frag(s_tgb, 64, rt * 16, k0, lane);
        v16bf b = load_b_frag(candT, 64, ct * 16, k0, lane);
        acc = wmma_bf16(a, b, acc);
      }
      int n  = ct * 16 + (lane & 15);
      int mb = rt * 16 + ((lane >> 4) << 3);
      if (n < NZ_) {
#pragma unroll
        for (int r = 0; r < 8; ++r)
          out_logits[(row0 + mb + r) * (size_t)NZ_ + n] = acc[r];
      }
    }
  }
  // purpose head: l2norm(pred_p[..,64:96]) @ Wdp + bdp  (reuse s_w as staging)
  float* s_pf = s_w;
  __syncthreads();
  for (int idx = tid; idx < 64 * 32; idx += 256) {
    int i = idx >> 5, k = idx & 31;
    s_pf[idx] = predp[(row0 + i) * 96 + 64 + k];
  }
  __syncthreads();
  if (tid < 64) {
    float ss = 0.f;
    for (int k = 0; k < 32; ++k) { float v = s_pf[tid * 32 + k]; ss += v * v; }
    s_rs[tid] = 1.0f / fmaxf(sqrtf(ss), 1e-12f);
  }
  __syncthreads();
  for (int idx = tid; idx < 64 * NPURP_; idx += 256) {
    int i = idx >> 3, j = idx & 7;
    float a = bdp[j];
    for (int k = 0; k < 32; ++k) a += s_pf[i * 32 + k] * s_rs[i] * Wdp[k * NPURP_ + j];
    out_purp[(row0 + i) * NPURP_ + j] = a;
  }
}

// ---------------- host launch ----------------
extern "C" void kernel_launch(void* const* d_in, const int* in_sizes, int n_in,
                              void* d_out, int out_size, void* d_ws, size_t ws_size,
                              hipStream_t stream) {
  const float* person = (const float*)d_in[0];
  const float* homez  = (const float*)d_in[1];
  const float* workz  = (const float*)d_in[2];
  const float* purp0  = (const float*)d_in[3];
  const float* times  = (const float*)d_in[4];
  const float* azf    = (const float*)d_in[5];
  const float* eps    = (const float*)d_in[6];
  const float* dW     = (const float*)d_in[7];
  const float* Wz     = (const float*)d_in[8];
  const float* bz     = (const float*)d_in[9];
  const float* We1    = (const float*)d_in[10];
  const float* be1    = (const float*)d_in[11];
  const float* We2    = (const float*)d_in[12];
  const float* be2    = (const float*)d_in[13];
  const float* Wn_in  = (const float*)d_in[14];
  const float* bn_in  = (const float*)d_in[15];
  const float* Wr1    = (const float*)d_in[16];
  const float* br1    = (const float*)d_in[17];
  const float* Wr2    = (const float*)d_in[18];
  const float* br2    = (const float*)d_in[19];
  const float* Wn_out = (const float*)d_in[20];
  const float* bn_out = (const float*)d_in[21];
  const float* Wdl    = (const float*)d_in[22];
  const float* bdl    = (const float*)d_in[23];
  const float* Wdp    = (const float*)d_in[24];
  const float* bdp    = (const float*)d_in[25];

  float* out = (float*)d_out;
  float* o_logits = out + OFF_LOGITS;
  float* o_loc    = out + OFF_LOC;
  float* o_purpl  = out + OFF_PURPL;
  float* o_purpf  = out + OFF_PURPF;
  float* o_p      = out + OFF_P;
  float* o_v      = out + OFF_V;
  float* o_mu     = out + OFF_MU;
  float* o_lv     = out + OFF_LV;

  char* ws = (char*)d_ws;
  bf16*  WnInT  = (bf16*)(ws + WS_WNIN);
  bf16*  Wr1T   = (bf16*)(ws + WS_WR1);
  bf16*  Wr2T   = (bf16*)(ws + WS_WR2);
  bf16*  WnOutT = (bf16*)(ws + WS_WNOUT);
  bf16*  candT  = (bf16*)(ws + WS_CAND);
  float* state0 = (float*)(ws + WS_STATE);
  bf16*  hbf    = (bf16*)(ws + WS_HBF);

  k_prep_weights<<<2048, 256, 0, stream>>>(Wn_in, Wr1, Wr2, Wn_out,
                                           WnInT, Wr1T, Wr2T, WnOutT);
  k_cand<<<NZP_, 64, 0, stream>>>(azf, Wz, bz, candT);
  k_encoder<<<B_, 256, 0, stream>>>(person, homez, workz, purp0, eps, Wz, bz,
                                    We1, be1, We2, be2, o_mu, o_lv, state0, hbf);
  k_scan<<<B_ / 16, 256, 0, stream>>>(times, dW, state0, hbf,
                                      WnInT, bn_in, Wr1T, br1, Wr2T, br2,
                                      WnOutT, bn_out, o_p, o_v, o_loc, o_purpf);
  k_decoder<<<(int)(NROWS / 64), 256, 0, stream>>>(o_p, Wdl, bdl, candT,
                                                   Wdp, bdp, o_logits, o_purpl);
}
